// MVMP_86122684220180
// MI455X (gfx1250) — matrix operand order repaired and checked by reading, not compiled, not run on hardware
//
#include <hip/hip_runtime.h>
#include <hip/hip_bf16.h>
#include <stdint.h>

#define HIDN 300
#define HEADS 4
#define DKH 75
#define KP1 320          // padded K for 300
#define KP3 960          // padded K for 900 (3 x 320 segments)
#define NT 20            // N tiles: 320/16
#define NTPW 4           // N tiles per wave (register blocking)
#define NGRP 5           // NT / NTPW

static const int NAc = 50000, NPc = 15000, EAc = 200000, EPc = 60000, ECc = 50000;

typedef __attribute__((ext_vector_type(16))) __bf16 v16bf;
typedef __attribute__((ext_vector_type(8)))  float  v8f;
typedef __attribute__((ext_vector_type(4)))  unsigned int u32x4;
typedef __attribute__((ext_vector_type(8)))  int i32x8;
typedef __attribute__((ext_vector_type(4)))  int i32x4;

#ifndef __has_builtin
#define __has_builtin(x) 0
#endif
#if __has_builtin(__builtin_amdgcn_tensor_load_to_lds)
#define HAVE_TDM 1
#else
#define HAVE_TDM 0
#endif
#if __has_builtin(__builtin_amdgcn_s_wait_tensorcnt)
#define HAVE_TWAIT 1
#else
#define HAVE_TWAIT 0
#endif

struct U8 { uint4 lo, hi; };

__device__ __forceinline__ unsigned short f2bf(float f) {
  union { float f; unsigned int u; } c; c.f = f;
  unsigned int u = c.u;
  unsigned int r = u + 0x7FFFu + ((u >> 16) & 1u);   // round-nearest-even
  return (unsigned short)(r >> 16);
}

// ---- f32 [rows, srcCols] -> bf16 [rows, dstStride] at column dstOff, zero-padded to padCols
__global__ void k_cvt_pad(const float* __restrict__ src, unsigned short* __restrict__ dst,
                          int rows, int srcCols, int dstStride, int dstOff, int padCols) {
  long long total = (long long)rows * padCols;
  for (long long i = blockIdx.x * (long long)blockDim.x + threadIdx.x; i < total;
       i += (long long)gridDim.x * blockDim.x) {
    int r = (int)(i / padCols);
    int c = (int)(i % padCols);
    unsigned short v = 0;
    if (c < srcCols) v = f2bf(src[(long long)r * srcCols + c]);
    dst[(long long)r * dstStride + dstOff + c] = v;
  }
}

// ---- pack W [K,N=300] row-major f32 -> B-fragment-ordered bf16 [nt][kblk][lane][16]
__global__ void k_pack_w(const float* __restrict__ W, unsigned short* __restrict__ out,
                         int K, int N, int KP) {
  int nKblk = KP >> 5;
  long long total = (long long)NT * nKblk * 32 * 16;
  for (long long i = blockIdx.x * (long long)blockDim.x + threadIdx.x; i < total;
       i += (long long)gridDim.x * blockDim.x) {
    int e    = (int)(i & 15);
    int lane = (int)((i >> 4) & 31);
    long long t = i >> 9;
    int kb = (int)(t % nKblk);
    int nt = (int)(t / nKblk);
    int n  = nt * 16 + (lane & 15);
    int kp = kb * 32 + ((lane >> 4) ? 16 : 0) + e;
    int seg = kp / KP1, off = kp % KP1;
    int k = seg * HIDN + off;
    unsigned short v = 0;
    if (off < HIDN && k < K && n < N) v = f2bf(W[(long long)k * N + n]);
    out[i] = v;
  }
}

// ---- TDM warm: DMA packed-weight panel into LDS in <=16KB tiles (pulls it into L2)
__global__ void __launch_bounds__(32) k_tdm_warm(const unsigned short* __restrict__ src,
                                                 int bytes, float* __restrict__ sink) {
#if defined(__AMDGCN__) && HAVE_TDM
  __shared__ unsigned long long shbuf[2048];              // 16 KB staging tile
  unsigned int lds = (unsigned int)(size_t)&shbuf[0];
  unsigned long long base = (unsigned long long)(size_t)src;
  for (int off = 0; off < bytes; off += 16384) {
    int rem = bytes - off;
    int chunk = rem < 16384 ? rem : 16384;
    int ne = chunk >> 3;                                   // 8-byte elements
    unsigned long long ga = base + (unsigned)off;
    u32x4 g0;
    g0[0] = 1u;                                            // count=1, user descriptor
    g0[1] = lds;                                           // lds_addr
    g0[2] = (unsigned)ga;                                  // global_addr[31:0]
    g0[3] = (unsigned)((ga >> 32) & 0x1FFFFFFu) | (2u << 30); // addr[56:32] | type=2
    i32x8 g1;
    g1[0] = (int)(3u << 16);                               // mask=0, data_size=8B
    g1[1] = (int)((ne & 0xFFFF) << 16);                    // tensor_dim0 lo
    g1[2] = (int)(((ne >> 16) & 0xFFFF) | (1 << 16));      // dim0 hi | tensor_dim1=1
    g1[3] = (int)((ne & 0xFFFF) << 16);                    // tile_dim0 = ne
    g1[4] = 1;                                             // tile_dim1=1, tile_dim2=0
    g1[5] = ne;                                            // tensor_dim0_stride
    g1[6] = 0; g1[7] = 0;
    i32x4 gz = {0, 0, 0, 0};
#if __clang_major__ >= 23
    i32x8 g4z = {0, 0, 0, 0, 0, 0, 0, 0};
    __builtin_amdgcn_tensor_load_to_lds(g0, g1, gz, gz, g4z, 0);
#else
    __builtin_amdgcn_tensor_load_to_lds(g0, g1, gz, gz, 0);
#endif
  }
#if HAVE_TWAIT
  __builtin_amdgcn_s_wait_tensorcnt(0);
#endif
  __syncthreads();
  if (threadIdx.x == 0) sink[0] = (float)(shbuf[0] & 0xFFu);
#else
  (void)src; (void)bytes;
  if (threadIdx.x == 0) sink[0] = 0.f;
#endif
}

// ---- C[M,300] = A_bf16[M,KP] @ Wpack (+bias)(+resid)(+relu); NTPW n-tiles per wave,
//      optional fused second weight sharing the A stream (DUAL=1 -> also C1 = A@Wp1+b1)
template <int DUAL>
__global__ void __launch_bounds__(32)
k_gemm_wmma(const unsigned short* __restrict__ A,
            const unsigned short* __restrict__ Wp0, const unsigned short* __restrict__ Wp1,
            const float* __restrict__ bias0, const float* __restrict__ bias1,
            const float* __restrict__ resid, float* __restrict__ C0, float* __restrict__ C1,
            int M, int KP, int relu) {
  int lane = threadIdx.x;
  int l15  = lane & 15;
  int hi   = lane >> 4;
  int ng   = blockIdx.y;                       // n-tile group: tiles ng*NTPW..+NTPW-1
  int mrow = blockIdx.x * 16 + l15;
  if (mrow >= M) mrow = M - 1;                 // clamp; garbage lanes never stored
  int nKblk = KP >> 5;
  const unsigned short* Ar = A + (long long)mrow * KP;
  v8f z = {0.f, 0.f, 0.f, 0.f, 0.f, 0.f, 0.f, 0.f};
  v8f acc0[NTPW] = {z, z, z, z};
  v8f acc1[NTPW] = {z, z, z, z};
  for (int kb = 0; kb < nKblk; ++kb) {
    U8 au;
    au.lo = *(const uint4*)(Ar + kb * 32 + hi * 16);       // K: kb*32 + 16h + s, s=0..7
    au.hi = *(const uint4*)(Ar + kb * 32 + hi * 16 + 8);   // K: kb*32 + 16h + s, s=8..15
    v16bf av = __builtin_bit_cast(v16bf, au);
#pragma unroll
    for (int t = 0; t < NTPW; ++t) {
      long long nt = (long long)(ng * NTPW + t);
      const unsigned short* bl0 = Wp0 + ((nt * nKblk + kb) << 9) + lane * 16;
      U8 bu;
      bu.lo = *(const uint4*)(bl0);
      bu.hi = *(const uint4*)(bl0 + 8);
      v16bf bv = __builtin_bit_cast(v16bf, bu);
      acc0[t] = __builtin_amdgcn_wmma_f32_16x16x32_bf16(false, av, false, bv,
                                                        (short)0, acc0[t], false, false);
      if (DUAL) {
        const unsigned short* bl1 = Wp1 + ((nt * nKblk + kb) << 9) + lane * 16;
        U8 cu;
        cu.lo = *(const uint4*)(bl1);
        cu.hi = *(const uint4*)(bl1 + 8);
        v16bf cv = __builtin_bit_cast(v16bf, cu);
        acc1[t] = __builtin_amdgcn_wmma_f32_16x16x32_bf16(false, av, false, cv,
                                                          (short)0, acc1[t], false, false);
      }
    }
  }
#pragma unroll
  for (int t = 0; t < NTPW; ++t) {
    int col = (ng * NTPW + t) * 16 + l15;
    if (col < HIDN) {
      float b0 = bias0 ? bias0[col] : 0.f;
      float b1 = (DUAL && bias1) ? bias1[col] : 0.f;
#pragma unroll
      for (int r = 0; r < 8; ++r) {
        int orow = blockIdx.x * 16 + hi * 8 + r;           // C/D VGPR layout
        if (orow < M) {
          float v = acc0[t][r] + b0;
          if (resid) v += resid[(long long)orow * HIDN + col];
          if (relu)  v = fmaxf(v, 0.f);
          C0[(long long)orow * HIDN + col] = v;
          if (DUAL) C1[(long long)orow * HIDN + col] = acc1[t][r] + b1;
        }
      }
    }
  }
}

__global__ void k_fill(float* __restrict__ p, long long n, float v) {
  for (long long i = blockIdx.x * (long long)blockDim.x + threadIdx.x; i < n;
       i += (long long)gridDim.x * blockDim.x) p[i] = v;
}

__global__ void k_attn_score(const float* __restrict__ q, const float* __restrict__ k,
                             const int* __restrict__ dst, float* __restrict__ s, int E) {
  long long total = (long long)E * HEADS;
  for (long long i = blockIdx.x * (long long)blockDim.x + threadIdx.x; i < total;
       i += (long long)gridDim.x * blockDim.x) {
    int e = (int)(i >> 2), h = (int)(i & 3);
    const float* qr = q + (long long)dst[e] * HIDN + h * DKH;
    const float* kr = k + (long long)e * HIDN + h * DKH;
    float acc = 0.f;
    for (int d = 0; d < DKH; ++d) acc += qr[d] * kr[d];
    s[i] = acc * 0.11547005383792515f;                      // 1/sqrt(75)
  }
}

__device__ __forceinline__ void atomicMaxF(float* addr, float val) {
  unsigned int* ua = (unsigned int*)addr;
  unsigned int old = *ua;
  while (__uint_as_float(old) < val) {
    unsigned int assumed = old;
    old = atomicCAS(ua, assumed, __float_as_uint(val));
    if (old == assumed) break;
  }
}

__global__ void k_segmax(const float* __restrict__ s, const int* __restrict__ dst,
                         float* __restrict__ smax, int E) {
  long long total = (long long)E * HEADS;
  for (long long i = blockIdx.x * (long long)blockDim.x + threadIdx.x; i < total;
       i += (long long)gridDim.x * blockDim.x)
    atomicMaxF(&smax[(long long)dst[i >> 2] * HEADS + (i & 3)], s[i]);
}

__global__ void k_finite0(float* __restrict__ p, long long n) {
  for (long long i = blockIdx.x * (long long)blockDim.x + threadIdx.x; i < n;
       i += (long long)gridDim.x * blockDim.x)
    if (!__builtin_isfinite(p[i])) p[i] = 0.f;
}

__global__ void k_expdenom(const float* __restrict__ s, const int* __restrict__ dst,
                           const float* __restrict__ smax, float* __restrict__ ebuf,
                           float* __restrict__ denom, int E) {
  long long total = (long long)E * HEADS;
  for (long long i = blockIdx.x * (long long)blockDim.x + threadIdx.x; i < total;
       i += (long long)gridDim.x * blockDim.x) {
    int e = (int)(i >> 2), h = (int)(i & 3);
    float v = __expf(s[i] - smax[(long long)dst[e] * HEADS + h]);
    ebuf[i] = v;
    atomicAdd(&denom[(long long)dst[e] * HEADS + h], v);
  }
}

__global__ void k_ctx(const float* __restrict__ ebuf, const float* __restrict__ denom,
                      const float* __restrict__ v, const int* __restrict__ dst,
                      float* __restrict__ ctx, int E) {
  long long total = (long long)E * HIDN;
  for (long long i = blockIdx.x * (long long)blockDim.x + threadIdx.x; i < total;
       i += (long long)gridDim.x * blockDim.x) {
    int e = (int)(i / HIDN), d = (int)(i % HIDN), h = d / DKH;
    int n = dst[e];
    float den = denom[(long long)n * HEADS + h];
    den = den > 0.f ? den : 1.f;
    float p = ebuf[(long long)e * HEADS + h] / den;
    atomicAdd(&ctx[(long long)n * HIDN + d], p * v[i]);
  }
}

__global__ void k_deg(const int* __restrict__ dst, float* __restrict__ deg, int E) {
  for (long long i = blockIdx.x * (long long)blockDim.x + threadIdx.x; i < E;
       i += (long long)gridDim.x * blockDim.x) atomicAdd(&deg[dst[i]], 1.f);
}

__global__ void k_select(const float* __restrict__ nw, const float* __restrict__ old,
                         const float* __restrict__ deg, float* __restrict__ out, int N) {
  long long total = (long long)N * HIDN;
  for (long long i = blockIdx.x * (long long)blockDim.x + threadIdx.x; i < total;
       i += (long long)gridDim.x * blockDim.x) {
    int n = (int)(i / HIDN);
    out[i] = deg[n] > 0.f ? nw[i] : old[i];
  }
}

// m = fa[src[e]] - h[e^1]  -> written directly as padded bf16 rows
__global__ void k_msg_bf(const float* __restrict__ fa, const float* __restrict__ h,
                         const int* __restrict__ src, unsigned short* __restrict__ dst, int E) {
  long long total = (long long)E * KP1;
  for (long long i = blockIdx.x * (long long)blockDim.x + threadIdx.x; i < total;
       i += (long long)gridDim.x * blockDim.x) {
    int e = (int)(i / KP1), c = (int)(i % KP1);
    unsigned short v = 0;
    if (c < HIDN)
      v = f2bf(fa[(long long)src[e] * HIDN + c] - h[(long long)(e ^ 1) * HIDN + c]);
    dst[i] = v;
  }
}

__global__ void k_scatter_add(const float* __restrict__ srcF, const int* __restrict__ sIdx,
                              const int* __restrict__ dIdx, float* __restrict__ out, int E) {
  long long total = (long long)E * HIDN;
  for (long long i = blockIdx.x * (long long)blockDim.x + threadIdx.x; i < total;
       i += (long long)gridDim.x * blockDim.x) {
    int e = (int)(i / HIDN), d = (int)(i % HIDN);
    int sr = sIdx ? sIdx[e] : e;
    atomicAdd(&out[(long long)dIdx[e] * HIDN + d], srcF[(long long)sr * HIDN + d]);
  }
}

static dim3 g1(long long n) {
  long long g = (n + 255) / 256;
  if (g > (1 << 20)) g = 1 << 20;
  if (g < 1) g = 1;
  return dim3((unsigned)g);
}

extern "C" void kernel_launch(void* const* d_in, const int* in_sizes, int n_in,
                              void* d_out, int out_size, void* d_ws, size_t ws_size,
                              hipStream_t stream) {
  const float* f_a  = (const float*)d_in[0];
  const float* f_p  = (const float*)d_in[1];
  const float* f_ja = (const float*)d_in[2];
  const float* f_jp = (const float*)d_in[3];
  const float* x_aba = (const float*)d_in[4];
  const float* x_prp = (const float*)d_in[5];
  const float* Wq_a = (const float*)d_in[6];  const float* bq_a = (const float*)d_in[7];
  const float* Wk_a = (const float*)d_in[8];  const float* bk_a = (const float*)d_in[9];
  const float* Wv_a = (const float*)d_in[10]; const float* bv_a = (const float*)d_in[11];
  const float* Wo_a = (const float*)d_in[12]; const float* bo_a = (const float*)d_in[13];
  const float* mpW_a = (const float*)d_in[14]; const float* mpb_a = (const float*)d_in[15];
  const float* Wq_p = (const float*)d_in[16]; const float* bq_p = (const float*)d_in[17];
  const float* Wk_p = (const float*)d_in[18]; const float* bk_p = (const float*)d_in[19];
  const float* Wv_p = (const float*)d_in[20]; const float* bv_p = (const float*)d_in[21];
  const float* Wo_p = (const float*)d_in[22]; const float* bo_p = (const float*)d_in[23];
  const float* mpW_p = (const float*)d_in[24]; const float* mpb_p = (const float*)d_in[25];
  const float* Wlast_a = (const float*)d_in[26]; const float* blast_a = (const float*)d_in[27];
  const float* Wlast_p = (const float*)d_in[28]; const float* blast_p = (const float*)d_in[29];
  const int* a_src = (const int*)d_in[30]; const int* a_dst = (const int*)d_in[31];
  const int* p_src = (const int*)d_in[32]; const int* p_dst = (const int*)d_in[33];
  const int* ap_src = (const int*)d_in[34]; const int* ap_dst = (const int*)d_in[35];
  const int* pa_src = (const int*)d_in[36]; const int* pa_dst = (const int*)d_in[37];

  // ---- workspace bump allocator
  char* ws = (char*)d_ws;
  size_t off = 0;
  auto alloc = [&](size_t bytes) -> void* {
    off = (off + 255) & ~(size_t)255;
    void* p = ws + off; off += bytes; return p;
  };
  const size_t pw300 = (size_t)NT * 10 * 512 * sizeof(unsigned short);
  const size_t pw900 = (size_t)NT * 30 * 512 * sizeof(unsigned short);
  unsigned short* PWq_a = (unsigned short*)alloc(pw300);
  unsigned short* PWk_a = (unsigned short*)alloc(pw300);
  unsigned short* PWv_a = (unsigned short*)alloc(pw300);
  unsigned short* PWo_a = (unsigned short*)alloc(pw300);
  unsigned short* PWq_p = (unsigned short*)alloc(pw300);
  unsigned short* PWk_p = (unsigned short*)alloc(pw300);
  unsigned short* PWv_p = (unsigned short*)alloc(pw300);
  unsigned short* PWo_p = (unsigned short*)alloc(pw300);
  unsigned short* PWmp_a[2] = {(unsigned short*)alloc(pw300), (unsigned short*)alloc(pw300)};
  unsigned short* PWmp_p[2] = {(unsigned short*)alloc(pw300), (unsigned short*)alloc(pw300)};
  unsigned short* PWl_a = (unsigned short*)alloc(pw900);
  unsigned short* PWl_p = (unsigned short*)alloc(pw900);
  unsigned short* bfNode = (unsigned short*)alloc((size_t)NAc * KP1 * 2);
  unsigned short* bfEdge = (unsigned short*)alloc((size_t)EAc * KP1 * 2);
  unsigned short* bfCat  = (unsigned short*)alloc((size_t)NAc * KP3 * 2);
  float* qN   = (float*)alloc((size_t)NAc * HIDN * 4);
  float* kE   = (float*)alloc((size_t)EAc * HIDN * 4);
  float* vE   = (float*)alloc((size_t)EAc * HIDN * 4);
  float* ctx  = (float*)alloc((size_t)NAc * HIDN * 4);
  float* attO = (float*)alloc((size_t)NAc * HIDN * 4);
  float* sE   = (float*)alloc((size_t)EAc * HEADS * 4);
  float* eE   = (float*)alloc((size_t)EAc * HEADS * 4);
  float* smax = (float*)alloc((size_t)NAc * HEADS * 4);
  float* den  = (float*)alloc((size_t)NAc * HEADS * 4);
  float* faA = (float*)alloc((size_t)NAc * HIDN * 4);
  float* faB = (float*)alloc((size_t)NAc * HIDN * 4);
  float* fpA = (float*)alloc((size_t)NPc * HIDN * 4);
  float* fpB = (float*)alloc((size_t)NPc * HIDN * 4);
  float* haA = (float*)alloc((size_t)EAc * HIDN * 4);
  float* haB = (float*)alloc((size_t)EAc * HIDN * 4);
  float* hpA = (float*)alloc((size_t)EPc * HIDN * 4);
  float* hpB = (float*)alloc((size_t)EPc * HIDN * 4);
  float* degA  = (float*)alloc((size_t)NAc * 4);
  float* degP  = (float*)alloc((size_t)NPc * 4);
  float* degAP = (float*)alloc((size_t)NPc * 4);
  float* degPA = (float*)alloc((size_t)NAc * 4);
  float* mail = (float*)alloc((size_t)NAc * HIDN * 4);
  float* tmpN = (float*)alloc((size_t)NAc * HIDN * 4);
  float* sinkF = (float*)alloc(64 * 4);
  (void)ws_size; (void)in_sizes; (void)n_in; (void)out_size;

  const int TB = 256;
  auto fill = [&](float* p, long long n, float v) { k_fill<<<g1(n), TB, 0, stream>>>(p, n, v); };
  auto pack = [&](const float* W, unsigned short* out, int K, int KP) {
    long long n = (long long)NT * (KP >> 5) * 512;
    k_pack_w<<<g1(n), TB, 0, stream>>>(W, out, K, HIDN, KP);
  };
  auto cvt = [&](const float* src, unsigned short* dst, int rows, int dstStride, int dstOff) {
    k_cvt_pad<<<g1((long long)rows * KP1), TB, 0, stream>>>(src, dst, rows, HIDN, dstStride, dstOff, KP1);
  };
  auto gemm1 = [&](const unsigned short* A, const unsigned short* Wp, const float* bias,
                   const float* resid, float* C, int M, int KP, int relu) {
    dim3 grid((M + 15) / 16, NGRP);
    k_gemm_wmma<0><<<grid, 32, 0, stream>>>(A, Wp, nullptr, bias, nullptr, resid, C, nullptr, M, KP, relu);
  };
  auto gemm2 = [&](const unsigned short* A, const unsigned short* Wp0, const unsigned short* Wp1,
                   const float* b0, const float* b1, float* C0, float* C1, int M) {
    dim3 grid((M + 15) / 16, NGRP);
    k_gemm_wmma<1><<<grid, 32, 0, stream>>>(A, Wp0, Wp1, b0, b1, nullptr, C0, C1, M, KP1, 0);
  };
  auto scat = [&](const float* s, const int* si, const int* di, float* o, int E) {
    k_scatter_add<<<g1((long long)E * HIDN), TB, 0, stream>>>(s, si, di, o, E);
  };
  int warmIdx = 0;
  auto warm = [&](const unsigned short* p, size_t bytes) {
    k_tdm_warm<<<1, 32, 0, stream>>>(p, (int)bytes, sinkF + (warmIdx++ & 63));
  };

  // ---- weight packing (deterministic, every call) + TDM warm of packed panels
  pack(Wq_a, PWq_a, HIDN, KP1); pack(Wk_a, PWk_a, HIDN, KP1);
  pack(Wv_a, PWv_a, HIDN, KP1); pack(Wo_a, PWo_a, HIDN, KP1);
  pack(Wq_p, PWq_p, HIDN, KP1); pack(Wk_p, PWk_p, HIDN, KP1);
  pack(Wv_p, PWv_p, HIDN, KP1); pack(Wo_p, PWo_p, HIDN, KP1);
  pack(mpW_a,               PWmp_a[0], HIDN, KP1);
  pack(mpW_a + HIDN * HIDN, PWmp_a[1], HIDN, KP1);
  pack(mpW_p,               PWmp_p[0], HIDN, KP1);
  pack(mpW_p + HIDN * HIDN, PWmp_p[1], HIDN, KP1);
  pack(Wlast_a, PWl_a, 3 * HIDN, KP3);
  pack(Wlast_p, PWl_p, 3 * HIDN, KP3);
  warm(PWq_a, pw300); warm(PWk_a, pw300); warm(PWv_a, pw300); warm(PWo_a, pw300);
  warm(PWq_p, pw300); warm(PWk_p, pw300); warm(PWv_p, pw300); warm(PWo_p, pw300);
  warm(PWmp_a[0], pw300); warm(PWmp_a[1], pw300);
  warm(PWmp_p[0], pw300); warm(PWmp_p[1], pw300);
  warm(PWl_a, pw900); warm(PWl_p, pw900);

  // ---- degrees
  fill(degA, NAc, 0.f); fill(degP, NPc, 0.f); fill(degAP, NPc, 0.f); fill(degPA, NAc, 0.f);
  k_deg<<<g1(EAc), TB, 0, stream>>>(a_dst, degA, EAc);
  k_deg<<<g1(EPc), TB, 0, stream>>>(p_dst, degP, EPc);
  k_deg<<<g1(ECc), TB, 0, stream>>>(ap_dst, degAP, ECc);
  k_deg<<<g1(ECc), TB, 0, stream>>>(pa_dst, degPA, ECc);

  // ---- state init
  hipMemcpyAsync(faA, f_a, (size_t)NAc * HIDN * 4, hipMemcpyDeviceToDevice, stream);
  hipMemcpyAsync(fpA, f_p, (size_t)NPc * HIDN * 4, hipMemcpyDeviceToDevice, stream);
  hipMemcpyAsync(haA, x_aba, (size_t)EAc * HIDN * 4, hipMemcpyDeviceToDevice, stream);
  hipMemcpyAsync(hpA, x_prp, (size_t)EPc * HIDN * 4, hipMemcpyDeviceToDevice, stream);

  auto attn = [&](float* feat, float* hcur, const int* dstIdx, int N, int E,
                  unsigned short* PWq, const float* bq, unsigned short* PWk, const float* bk,
                  unsigned short* PWv, const float* bv, unsigned short* PWo, const float* bo,
                  const float* deg, float* featOut) {
    cvt(feat, bfNode, N, KP1, 0);
    gemm1(bfNode, PWq, bq, nullptr, qN, N, KP1, 0);
    cvt(hcur, bfEdge, E, KP1, 0);
    gemm2(bfEdge, PWk, PWv, bk, bv, kE, vE, E);            // fused k+v: A streamed once
    k_attn_score<<<g1((long long)E * HEADS), TB, 0, stream>>>(qN, kE, dstIdx, sE, E);
    fill(smax, (long long)N * HEADS, -__builtin_inff());
    k_segmax<<<g1((long long)E * HEADS), TB, 0, stream>>>(sE, dstIdx, smax, E);
    k_finite0<<<g1((long long)N * HEADS), TB, 0, stream>>>(smax, (long long)N * HEADS);
    fill(den, (long long)N * HEADS, 0.f);
    k_expdenom<<<g1((long long)E * HEADS), TB, 0, stream>>>(sE, dstIdx, smax, eE, den, E);
    fill(ctx, (long long)N * HIDN, 0.f);
    k_ctx<<<g1((long long)E * HIDN), TB, 0, stream>>>(eE, den, vE, dstIdx, ctx, E);
    cvt(ctx, bfNode, N, KP1, 0);
    gemm1(bfNode, PWo, bo, feat, attO, N, KP1, 0);
    k_select<<<g1((long long)N * HIDN), TB, 0, stream>>>(attO, feat, deg, featOut, N);
  };
  auto msg = [&](const float* faNew, const float* hcur, const float* x, const int* srcIdx,
                 int E, unsigned short* PWmp, const float* bmp, float* hNext) {
    k_msg_bf<<<g1((long long)E * KP1), TB, 0, stream>>>(faNew, hcur, srcIdx, bfEdge, E);
    gemm1(bfEdge, PWmp, bmp, x, hNext, E, KP1, 1);
  };

  float *faC = faA, *faN = faB, *fpC = fpA, *fpN = fpB;
  float *haC = haA, *haN = haB, *hpC = hpA, *hpN = hpB;
  for (int it = 0; it < 2; ++it) {
    attn(faC, haC, a_dst, NAc, EAc, PWq_a, bq_a, PWk_a, bk_a, PWv_a, bv_a, PWo_a, bo_a, degA, faN);
    attn(fpC, hpC, p_dst, NPc, EPc, PWq_p, bq_p, PWk_p, bk_p, PWv_p, bv_p, PWo_p, bo_p, degP, fpN);
    msg(faN, haC, x_aba, a_src, EAc, PWmp_a[it], mpb_a + it * HIDN, haN);
    msg(fpN, hpC, x_prp, p_src, EPc, PWmp_p[it], mpb_p + it * HIDN, hpN);
    float* t;
    t = faC; faC = faN; faN = t;  t = fpC; fpC = fpN; fpN = t;
    t = haC; haC = haN; haN = t;  t = hpC; hpC = hpN; hpN = t;
  }

  float* out_fa  = (float*)d_out;
  float* out_fp  = out_fa + (size_t)NAc * HIDN;
  float* out_fja = out_fp + (size_t)NPc * HIDN;
  float* out_fjp = out_fja + (size_t)NAc * HIDN;

  auto lastpass = [&](const float* mailBuf, const float* fcur, const float* f0,
                      unsigned short* PWl, const float* bl, const float* deg,
                      int N, float* outPtr) {
    cvt(mailBuf, bfCat, N, KP3, 0);
    cvt(fcur,    bfCat, N, KP3, KP1);
    cvt(f0,      bfCat, N, KP3, 2 * KP1);
    gemm1(bfCat, PWl, bl, nullptr, tmpN, N, KP3, 0);
    k_select<<<g1((long long)N * HIDN), TB, 0, stream>>>(tmpN, fcur, deg, outPtr, N);
  };

  // homo last pass
  fill(mail, (long long)NAc * HIDN, 0.f);
  scat(haC, nullptr, a_dst, mail, EAc);
  lastpass(mail, faC, f_a, PWl_a, blast_a, degA, NAc, out_fa);
  fill(mail, (long long)NPc * HIDN, 0.f);
  scat(hpC, nullptr, p_dst, mail, EPc);
  lastpass(mail, fpC, f_p, PWl_p, blast_p, degP, NPc, out_fp);
  // hetero junction pass (reads pre-update f_junc on both sides)
  fill(mail, (long long)NPc * HIDN, 0.f);
  scat(f_ja, ap_src, ap_dst, mail, ECc);
  lastpass(mail, f_jp, f_p, PWl_p, blast_p, degAP, NPc, out_fjp);
  fill(mail, (long long)NAc * HIDN, 0.f);
  scat(f_jp, pa_src, pa_dst, mail, ECc);
  lastpass(mail, f_ja, f_a, PWl_a, blast_a, degPA, NAc, out_fja);
}
